// MCS_HGPSL_65730179498545
// MI455X (gfx1250) — matrix operand order repaired and checked by reading, not compiled, not run on hardware
//
#include <hip/hip_runtime.h>
#include <hip/hip_bf16.h>
#include <math.h>
#include <stdint.h>

typedef _Float16 h16;
typedef __attribute__((ext_vector_type(16))) _Float16 v16h;
typedef __attribute__((ext_vector_type(8)))  _Float16 v8h;
typedef __attribute__((ext_vector_type(8)))  float    v8f;
typedef int vsi4 __attribute__((vector_size(16)));

#define NG 64
#define NF 128
#define KC 32

#if defined(__HIP_DEVICE_COMPILE__) && defined(__has_builtin)
#  if __has_builtin(__builtin_amdgcn_global_load_async_to_lds_b128) && \
      __has_builtin(__builtin_amdgcn_s_wait_asynccnt)
#    define USE_ASYNC_LDS 1
#  endif
#endif
#ifndef USE_ASYNC_LDS
#  define USE_ASYNC_LDS 0
#endif

#define AS1 __attribute__((address_space(1)))
#define AS3 __attribute__((address_space(3)))

// ---------------------------------------------------------------------------
// Batched WMMA GEMM: C[g](M x 128) = A[g](M x K) * B[g](K x 128) (+bias,+relu)
// f16 inputs, f32 accumulate.  Block = 256 threads (8 waves); each wave owns a
// 16-row strip of C (8 accumulators covering all 128 cols).  The B panel is
// staged per-KC-chunk into LDS once per block (async-to-LDS when available)
// and shared by all 8 waves.  grid = (ceil(Mtiles/8), G).
// Fragment layouts per CDNA5 ISA 7.12.2:
//   A 16x32 f16 : lane l -> row (l&15); elem e -> K = (e/8)*16 + (l>>4)*8 + e%8
//   B 32x16 f16 : lane l -> col (l&15); elem e -> K as for A
//   C 16x16 f32 : lane l -> col (l&15); elem e -> row e + (l>>4)*8
// ---------------------------------------------------------------------------
__global__ __launch_bounds__(256)
void wmma_gemm(const h16* __restrict__ A, const h16* __restrict__ B,
               float* __restrict__ C,
               int lda, int ldb, int ldc,
               long gsA, long gsB, long gsC,
               int K, int Mtiles, const float* __restrict__ bias, int relu)
{
    __shared__ h16 Bsh[KC][NF + 8];            // 272B row stride: aligned, low-conflict

    const int g    = blockIdx.y;
    const int t    = threadIdx.x;
    const int wave = t >> 5;
    const int lane = t & 31;
    const int hf   = lane >> 4;
    const int l15  = lane & 15;
    const int tm   = blockIdx.x * 8 + wave;
    const bool act = (tm < Mtiles);

    const h16* Ag = A + (long)g * gsA + (long)(tm * 16 + l15) * lda;
    const h16* Bg = B + (long)g * gsB;
    float*     Cg = C + (long)g * gsC;

    v8f acc[8];
#pragma unroll
    for (int i = 0; i < 8; ++i) acc[i] = (v8f){};

    for (int k0 = 0; k0 < K; k0 += KC) {
        // ---- cooperative stage of B[k0..k0+KC) x 128 into LDS (8 KB) ----
        {
            const int idx = t * 2;             // 512 x 16B pieces, 2 per thread
#pragma unroll
            for (int p = 0; p < 2; ++p) {
                const int id  = idx + p;
                const int row = id >> 4;       // 16 pieces per 128-half row
                const int col = (id & 15) * 8;
                const h16* src = Bg + (long)(k0 + row) * ldb + col;
#if USE_ASYNC_LDS
                __builtin_amdgcn_global_load_async_to_lds_b128(
                    (AS1 vsi4*)(uintptr_t)src,
                    (AS3 vsi4*)(uint32_t)(uintptr_t)&Bsh[row][col],
                    0, 0);
#else
                *(v8h*)&Bsh[row][col] = *(const v8h*)src;
#endif
            }
            if (k0 + KC < K) {                 // prefetch next chunk (global_prefetch_b8)
                const h16* nsrc = Bg + (long)(k0 + KC + (idx >> 4)) * ldb;
                __builtin_prefetch(nsrc, 0, 1);
            }
        }
#if USE_ASYNC_LDS
        __builtin_amdgcn_s_wait_asynccnt(0);
#endif
        __syncthreads();

        if (act) {
            v8h alo = *(const v8h*)(Ag + k0 + hf * 8);
            v8h ahi = *(const v8h*)(Ag + k0 + 16 + hf * 8);
            v16h af;
#pragma unroll
            for (int e = 0; e < 8; ++e) { af[e] = alo[e]; af[e + 8] = ahi[e]; }
#pragma unroll
            for (int tn = 0; tn < 8; ++tn) {
                v16h bf;
#pragma unroll
                for (int e = 0; e < 8; ++e) { bf[e] = Bsh[hf * 8 + e][tn * 16 + l15]; bf[e + 8] = Bsh[16 + hf * 8 + e][tn * 16 + l15]; }
                acc[tn] = __builtin_amdgcn_wmma_f32_16x16x32_f16(
                    false, af, false, bf, (short)0, acc[tn], false, false);
            }
        }
        __syncthreads();
    }

    if (act) {
#pragma unroll
        for (int tn = 0; tn < 8; ++tn) {
            const int col = tn * 16 + l15;
            const float bv = bias ? bias[col] : 0.0f;
#pragma unroll
            for (int e = 0; e < 8; ++e) {
                const int row = tm * 16 + hf * 8 + e;
                float v = acc[tn][e] + bv;
                if (relu) v = fmaxf(v, 0.0f);
                Cg[(long)row * ldc + col] = v;
            }
        }
    }
}

// f32 -> f16 elementwise
__global__ void cvt_f16(h16* __restrict__ dst, const float* __restrict__ src, int n)
{
    int i = blockIdx.x * 256 + threadIdx.x;
    if (i < n) dst[i] = (h16)src[i];
}

// per (g,i): deg (no self loop, clamped) and dinv = rsqrt(rowsum + 1)
__global__ void rowsum_kernel(const float* __restrict__ A, int ldA, long gsA, int n,
                              float* __restrict__ dinv, float* __restrict__ deg)
{
    int g = blockIdx.y;
    int i = blockIdx.x * 256 + threadIdx.x;
    if (i >= n) return;
    const float* row = A + (long)g * gsA + (long)i * ldA;
    float s = 0.0f;
    for (int j = 0; j < n; ++j) s += row[j];
    deg[g * 512 + i]  = fmaxf(s, 1e-8f);
    dinv[g * 512 + i] = rsqrtf(fmaxf(s + 1.0f, 1e-8f));
}

// An = D^-1/2 (A+I) D^-1/2 as f16 into zero-padded np x np buffer
__global__ void an_fill(const float* __restrict__ A, int ldA, long gsA, int n, int np,
                        const float* __restrict__ dinv, h16* __restrict__ out)
{
    int g = blockIdx.y;
    int e = blockIdx.x * 256 + threadIdx.x;
    if (e >= n * n) return;
    int i = e / n, j = e - i * n;
    float v = A[(long)g * gsA + (long)i * ldA + j] + (i == j ? 1.0f : 0.0f);
    v *= dinv[g * 512 + i] * dinv[g * 512 + j];
    out[(long)g * np * np + (long)i * np + j] = (h16)v;
}

// raw adjacency as f16 into zero-padded np x np buffer
__global__ void araw_fill(const float* __restrict__ A, int ldA, long gsA, int n, int np,
                          h16* __restrict__ out)
{
    int g = blockIdx.y;
    int e = blockIdx.x * 256 + threadIdx.x;
    if (e >= n * n) return;
    int i = e / n, j = e - i * n;
    out[(long)g * np * np + (long)i * np + j] = (h16)A[(long)g * gsA + (long)i * ldA + j];
}

// score[g][i] = sum_f |h - neigh/deg|
__global__ void score_kernel(const float* __restrict__ h, const float* __restrict__ neigh,
                             const float* __restrict__ deg, int n, long gs,
                             float* __restrict__ score)
{
    int g = blockIdx.y;
    int i = blockIdx.x * 256 + threadIdx.x;
    if (i >= n) return;
    float dv = 1.0f / deg[g * 512 + i];
    const float* hp  = h + (long)g * gs + (long)i * NF;
    const float* np_ = neigh + (long)g * gs + (long)i * NF;
    float s = 0.0f;
    for (int f = 0; f < NF; ++f) s += fabsf(hp[f] - np_[f] * dv);
    score[g * 512 + i] = s;
}

// top-k by selection; one block (256 thr) per graph, n <= 512, k <= 256
__global__ void topk_kernel(const float* __restrict__ score, int n, int k,
                            float* __restrict__ tops, int* __restrict__ idx)
{
    int g = blockIdx.x, t = threadIdx.x;
    __shared__ float s[512];
    __shared__ float rmax[256];
    __shared__ int   rarg[256];
    for (int i = t; i < n; i += 256) s[i] = score[g * 512 + i];
    __syncthreads();
    for (int sel = 0; sel < k; ++sel) {
        float m = -INFINITY; int a = 0x7fffffff;
        for (int i = t; i < n; i += 256) {
            float v = s[i];
            if (v > m || (v == m && i < a)) { m = v; a = i; }
        }
        rmax[t] = m; rarg[t] = a;
        __syncthreads();
        for (int o = 128; o > 0; o >>= 1) {
            if (t < o) {
                if (rmax[t + o] > rmax[t] ||
                    (rmax[t + o] == rmax[t] && rarg[t + o] < rarg[t])) {
                    rmax[t] = rmax[t + o]; rarg[t] = rarg[t + o];
                }
            }
            __syncthreads();
        }
        if (t == 0) {
            tops[g * 256 + sel] = rmax[0];
            idx[g * 256 + sel]  = rarg[0];
            s[rarg[0]] = -INFINITY;
        }
        __syncthreads();
    }
}

// hk[g][j][:] = h[g][idx[j]][:] * sigmoid(tops[j]);  grid (k, G), 128 threads
__global__ void gather_scale(const float* __restrict__ h, long gsh,
                             const int* __restrict__ idx, const float* __restrict__ tops,
                             int kp, float* __restrict__ hk)
{
    int g = blockIdx.y, j = blockIdx.x, f = threadIdx.x;
    float sg = 1.0f / (1.0f + expf(-tops[g * 256 + j]));
    int src = idx[g * 256 + j];
    hk[(long)g * kp * NF + (long)j * NF + f] =
        h[(long)g * gsh + (long)src * NF + f] * sg;
}

// si = hk @ a_src, sj = hk @ a_dst
__global__ void attn_kernel(const float* __restrict__ hk, int kp,
                            const float* __restrict__ a, int k,
                            float* __restrict__ si, float* __restrict__ sj)
{
    int g = blockIdx.y;
    int j = blockIdx.x * 128 + threadIdx.x;
    if (j >= k) return;
    const float* hp = hk + (long)g * kp * NF + (long)j * NF;
    float s1 = 0.0f, s2 = 0.0f;
    for (int f = 0; f < NF; ++f) { float v = hp[f]; s1 += v * a[f]; s2 += v * a[NF + f]; }
    si[g * 256 + j] = s1;
    sj[g * 256 + j] = s2;
}

// Anext[g][j][:] = softmax_l( relu(si_j + sj_l) + Ak[j][l] ); zero-pads cols to kp
__global__ void pool_softmax(const float* __restrict__ Acur, int ldA, long gsA,
                             const int* __restrict__ idx,
                             const float* __restrict__ si, const float* __restrict__ sj,
                             int k, int kp, float* __restrict__ Anext)
{
    int g = blockIdx.y, j = blockIdx.x, t = threadIdx.x;
    __shared__ float buf[256];
    __shared__ float red[256];
    int ri = idx[g * 256 + j];
    float v = -INFINITY;
    if (t < k) {
        int ci = idx[g * 256 + t];
        float lg = si[g * 256 + j] + sj[g * 256 + t];
        lg = lg > 0.0f ? lg : 0.0f;
        lg += Acur[(long)g * gsA + (long)ri * ldA + ci];  // LAMB = 1.0
        v = lg;
    }
    buf[t] = v; red[t] = v;
    __syncthreads();
    for (int o = 128; o > 0; o >>= 1) { if (t < o) red[t] = fmaxf(red[t], red[t + o]); __syncthreads(); }
    float mx = red[0];
    __syncthreads();
    float e = (t < k) ? expf(buf[t] - mx) : 0.0f;
    red[t] = e;
    __syncthreads();
    for (int o = 128; o > 0; o >>= 1) { if (t < o) red[t] += red[t + o]; __syncthreads(); }
    float sum = red[0];
    if (t < kp) Anext[(long)g * kp * kp + (long)j * kp + t] = e / sum;
}

// r[g] += [relu(max_j h), relu(mean_j h)]
__global__ void readout_acc(const float* __restrict__ h, int k, long gs,
                            float* __restrict__ r)
{
    int g = blockIdx.y, f = threadIdx.x;
    const float* hg = h + (long)g * gs;
    float mx = -INFINITY, sm = 0.0f;
    for (int j = 0; j < k; ++j) {
        float v = hg[(long)j * NF + f];
        mx = fmaxf(mx, v); sm += v;
    }
    r[g * 2 * NF + f]      += fmaxf(mx, 0.0f);
    r[g * 2 * NF + NF + f] += fmaxf(sm / (float)k, 0.0f);
}

__global__ void mlp_kernel(const float* __restrict__ in, const float* __restrict__ W,
                           const float* __restrict__ b, float* __restrict__ out,
                           int K, int O, int relu)
{
    int g = blockIdx.y;
    int o = blockIdx.x * blockDim.x + threadIdx.x;
    if (o >= O) return;
    float s = b[o];
    for (int c = 0; c < K; ++c) s += in[g * K + c] * W[c * O + o];
    if (relu) s = fmaxf(s, 0.0f);
    out[g * O + o] = s;
}

__global__ void logsoftmax_kernel(const float* __restrict__ z, const float* __restrict__ W,
                                  const float* __restrict__ b, float* __restrict__ out)
{
    int g = blockIdx.x, t = threadIdx.x;
    __shared__ float lg[10];
    if (t < 10) {
        float s = b[t];
        for (int c = 0; c < 64; ++c) s += z[g * 64 + c] * W[c * 10 + t];
        lg[t] = s;
    }
    __syncthreads();
    if (t < 10) {
        float mx = -INFINITY;
        for (int i = 0; i < 10; ++i) mx = fmaxf(mx, lg[i]);
        float sm = 0.0f;
        for (int i = 0; i < 10; ++i) sm += expf(lg[i] - mx);
        out[g * 10 + t] = lg[t] - mx - logf(sm);
    }
}

extern "C" void kernel_launch(void* const* d_in, const int* in_sizes, int n_in,
                              void* d_out, int out_size, void* d_ws, size_t ws_size,
                              hipStream_t stream)
{
    (void)in_sizes; (void)n_in; (void)out_size; (void)ws_size;

    const float* x     = (const float*)d_in[0];
    const float* adj   = (const float*)d_in[1];
    const float* Ws_[5] = { (const float*)d_in[2], (const float*)d_in[4],
                            (const float*)d_in[6], (const float*)d_in[8],
                            (const float*)d_in[6] };
    const float* bs_[5] = { (const float*)d_in[3], (const float*)d_in[5],
                            (const float*)d_in[7], (const float*)d_in[9],
                            (const float*)d_in[7] };
    const float* as_[4] = { (const float*)d_in[10], (const float*)d_in[11],
                            (const float*)d_in[12], (const float*)d_in[13] };
    const float* lin1W = (const float*)d_in[14]; const float* lin1b = (const float*)d_in[15];
    const float* lin2W = (const float*)d_in[16]; const float* lin2b = (const float*)d_in[17];
    const float* lin3W = (const float*)d_in[18]; const float* lin3b = (const float*)d_in[19];
    float* out = (float*)d_out;

    // ---- workspace carve ----
    size_t off = 0;
    char* base = (char*)d_ws;
    auto alloc = [&](size_t bytes) -> void* {
        void* p = base + off; off += (bytes + 255) & ~(size_t)255; return p;
    };
    h16*   adjh  = (h16*)  alloc((size_t)NG * 512 * 512 * 2);   // An / Araw (shared)
    float* Aar0  = (float*)alloc((size_t)NG * 256 * 256 * 4);
    float* Aar1  = (float*)alloc((size_t)NG * 256 * 256 * 4);
    float* H     = (float*)alloc((size_t)NG * 512 * NF * 4);    // gcn output
    float* HK    = (float*)alloc((size_t)NG * 512 * NF * 4);    // pooled features
    float* T     = (float*)alloc((size_t)NG * 512 * NF * 4);    // hW / neigh
    h16*   Hh    = (h16*)  alloc((size_t)NG * 512 * NF * 2);
    h16*   Th    = (h16*)  alloc((size_t)NG * 512 * NF * 2);
    h16*   Wh4[4];
    for (int i = 0; i < 4; ++i) Wh4[i] = (h16*)alloc((size_t)NF * NF * 2);
    float* dinv  = (float*)alloc((size_t)NG * 512 * 4);
    float* deg   = (float*)alloc((size_t)NG * 512 * 4);
    float* score = (float*)alloc((size_t)NG * 512 * 4);
    float* tops  = (float*)alloc((size_t)NG * 256 * 4);
    int*   idxb  = (int*)  alloc((size_t)NG * 256 * 4);
    float* si    = (float*)alloc((size_t)NG * 256 * 4);
    float* sj    = (float*)alloc((size_t)NG * 256 * 4);
    float* r     = (float*)alloc((size_t)NG * 256 * 4);
    float* z1    = (float*)alloc((size_t)NG * 128 * 4);
    float* z2    = (float*)alloc((size_t)NG * 64 * 4);

    h16* Whs[5] = { Wh4[0], Wh4[1], Wh4[2], Wh4[3], Wh4[2] };

    // weights -> f16 (W1, W2, W3, W5)
    const float* Wsrc[4] = { Ws_[0], Ws_[1], Ws_[2], Ws_[3] };
    for (int i = 0; i < 4; ++i)
        cvt_f16<<<(NF * NF + 255) / 256, 256, 0, stream>>>(Wh4[i], Wsrc[i], NF * NF);

    (void)hipMemsetAsync(r, 0, (size_t)NG * 256 * 4, stream);

    const int ns[5]  = { 512, 256, 128, 38, 11 };
    const int nps[5] = { 512, 256, 128, 64, 32 };
    const int kks[4] = { 256, 128, 38, 11 };
    const int kps[4] = { 256, 128, 64, 32 };

    const float* hcur = x;
    const float* Acur = adj;
    int  ldA = 512;
    long gsA = 512L * 512;
    float* Aarena[2] = { Aar0, Aar1 };

    for (int s = 0; s < 5; ++s) {
        const int n = ns[s], np = nps[s];
        const long gh = (long)np * NF;
        const int cnt = NG * np * NF;
        const int mt  = np / 16;                 // M tiles
        const dim3 ggrid((mt + 7) / 8, NG);

        // degree / normalization terms
        rowsum_kernel<<<dim3((n + 255) / 256, NG), 256, 0, stream>>>(Acur, ldA, gsA, n, dinv, deg);

        // h -> f16
        cvt_f16<<<(cnt + 255) / 256, 256, 0, stream>>>(Hh, hcur, cnt);

        // An (f16, zero padded)
        (void)hipMemsetAsync(adjh, 0, (size_t)NG * np * np * 2, stream);
        an_fill<<<dim3((n * n + 255) / 256, NG), 256, 0, stream>>>(Acur, ldA, gsA, n, np, dinv, adjh);

        // T = h @ W           (M=np, N=128, K=128)
        wmma_gemm<<<ggrid, 256, 0, stream>>>(
            Hh, Whs[s], T, NF, NF, NF, gh, 0L, gh, NF, mt, nullptr, 0);
        cvt_f16<<<(cnt + 255) / 256, 256, 0, stream>>>(Th, T, cnt);

        // H = relu(An @ T + b) (M=np, N=128, K=np)
        wmma_gemm<<<ggrid, 256, 0, stream>>>(
            adjh, Th, H, np, NF, NF, (long)np * np, gh, gh, np, mt, bs_[s], 1);

        if (s < 4) {
            const int k = kks[s], kp = kps[s];

            // raw A (f16, zero padded) for neighbor aggregation
            (void)hipMemsetAsync(adjh, 0, (size_t)NG * np * np * 2, stream);
            araw_fill<<<dim3((n * n + 255) / 256, NG), 256, 0, stream>>>(Acur, ldA, gsA, n, np, adjh);

            cvt_f16<<<(cnt + 255) / 256, 256, 0, stream>>>(Hh, H, cnt);

            // T = A @ H  (neighbor sums)
            wmma_gemm<<<ggrid, 256, 0, stream>>>(
                adjh, Hh, T, np, NF, NF, (long)np * np, gh, gh, np, mt, nullptr, 0);

            score_kernel<<<dim3((n + 255) / 256, NG), 256, 0, stream>>>(H, T, deg, n, gh, score);
            topk_kernel<<<NG, 256, 0, stream>>>(score, n, k, tops, idxb);
            gather_scale<<<dim3(k, NG), 128, 0, stream>>>(H, gh, idxb, tops, kp, HK);
            attn_kernel<<<dim3((k + 127) / 128, NG), 128, 0, stream>>>(HK, kp, as_[s], k, si, sj);
            pool_softmax<<<dim3(k, NG), 256, 0, stream>>>(
                Acur, ldA, gsA, idxb, si, sj, k, kp, Aarena[s & 1]);
            readout_acc<<<dim3(1, NG), NF, 0, stream>>>(HK, k, (long)kp * NF, r);

            hcur = HK;
            Acur = Aarena[s & 1];
            ldA  = kp;
            gsA  = (long)kp * kp;
        } else {
            readout_acc<<<dim3(1, NG), NF, 0, stream>>>(H, n, gh, r);
        }
    }

    // final MLP + log_softmax
    mlp_kernel<<<dim3(1, NG), 128, 0, stream>>>(r,  lin1W, lin1b, z1, 256, 128, 1);
    mlp_kernel<<<dim3(1, NG), 64,  0, stream>>>(z1, lin2W, lin2b, z2, 128, 64, 1);
    logsoftmax_kernel<<<NG, 32, 0, stream>>>(z2, lin3W, lin3b, out);
}
